// NonlinearReservoirCell_80839874445748
// MI455X (gfx1250) — compile-verified, matrix operand order checked by reading, not run on hardware
//
#include <hip/hip_runtime.h>

#define B_TOTAL 262144
#define M_L 3
#define N_F 256
#define H1 256
#define H2 128

typedef __attribute__((ext_vector_type(16))) _Float16 v16h;
typedef __attribute__((ext_vector_type(8)))  _Float16 v8h;
typedef __attribute__((ext_vector_type(8)))  float    v8f;
typedef __attribute__((ext_vector_type(4)))  float    v4f;

// ---------------------------------------------------------------------------
// Pre-pass: convert fp32 weights to f16, transposed to [m][out][in] so that a
// WMMA B-fragment is two contiguous 16B loads per lane (global_load_b128).
// ---------------------------------------------------------------------------
__global__ __launch_bounds__(256) void prep_weights(
    const float* __restrict__ W1, const float* __restrict__ W2,
    _Float16* __restrict__ W1t, _Float16* __restrict__ W2t)
{
    const int i  = blockIdx.x * 256 + threadIdx.x;
    const int n1 = M_L * N_F * H1;   // 196608
    const int n2 = M_L * H1 * H2;    //  98304
    if (i < n1) {
        const int m = i / (H1 * N_F);
        const int r = i % (H1 * N_F);
        const int h = r / N_F;       // output feature
        const int n = r % N_F;       // input feature
        W1t[i] = (_Float16)W1[(size_t)m * N_F * H1 + (size_t)n * H1 + h];
    } else if (i < n1 + n2) {
        const int j  = i - n1;
        const int m  = j / (H2 * H1);
        const int r  = j % (H2 * H1);
        const int h2 = r / H1;       // output feature
        const int k  = r % H1;       // input feature
        W2t[j] = (_Float16)W2[(size_t)m * H1 * H2 + (size_t)k * H2 + h2];
    }
}

// ---------------------------------------------------------------------------
// Main kernel: 16 batch rows per workgroup, 8 waves, loop over m.
// Layer1: 16x256 @ 256x256 (16 n-tiles, 2/wave, 8 k-steps of wmma 16x16x32)
// Layer2: 16x256 @ 256x128 (8 n-tiles, 1/wave)
// Layer3: per-row dot with W3, sigmoid blend, fp32 throughout the scalar path.
// ---------------------------------------------------------------------------
__global__ __launch_bounds__(256) void reservoir_kernel(
    const float* __restrict__ last_flow,
    const float* __restrict__ last_free_water,
    const float* __restrict__ runoff,
    const float* __restrict__ b1,
    const float* __restrict__ b2,
    const float* __restrict__ W3,
    const float* __restrict__ b3,
    const _Float16* __restrict__ W1t,
    const _Float16* __restrict__ W2t,
    float* __restrict__ out)
{
    __shared__ __align__(16) _Float16 sA[16 * N_F];  // 8KB: A tile, reused for a1
    __shared__ float sRow[16][16];                   // partial row sums / dot partials
    __shared__ float sP[16 * H2];                    // 8KB: layer-3 products
    __shared__ float sFW[16];                        // per-row runoff sums

    const int tid  = threadIdx.x;
    const int wave = tid >> 5;
    const int lane = tid & 31;
    const int hi   = lane >> 4;       // 0: K 0..7/16..23   1: K 8..15/24..31
    const int l16  = lane & 15;
    const int b0   = blockIdx.x * 16;
    const int lrow = tid >> 4;        // load-phase row 0..15
    const int lseg = tid & 15;        // load-phase 16-float segment

    for (int m = 0; m < M_L; ++m) {
        // ---- load A tile (coalesced b128 x4), fp32->f16, partial row sums ----
        {
            const float* src = runoff + (((size_t)(b0 + lrow)) * M_L + m) * N_F + lseg * 16;
            v4f f0 = *(const v4f*)(src + 0);
            v4f f1 = *(const v4f*)(src + 4);
            v4f f2 = *(const v4f*)(src + 8);
            v4f f3 = *(const v4f*)(src + 12);
            float s = 0.f;
            v8h hl, hh;
            #pragma unroll
            for (int j = 0; j < 4; ++j) {
                s += f0[j] + f1[j] + f2[j] + f3[j];
                hl[j]     = (_Float16)f0[j];
                hl[4 + j] = (_Float16)f1[j];
                hh[j]     = (_Float16)f2[j];
                hh[4 + j] = (_Float16)f3[j];
            }
            v8h* dst = (v8h*)&sA[lrow * N_F + lseg * 16];
            dst[0] = hl;
            dst[1] = hh;
            sRow[lrow][lseg] = s;
        }
        __syncthreads();
        if (tid < 16) {
            float s = 0.f;
            #pragma unroll
            for (int j = 0; j < 16; ++j) s += sRow[tid][j];
            sFW[tid] = s;   // sum(runoff[b, m, :])
        }

        // ---- layer 1: two 16x16 output tiles per wave ----
        v8f accL1[2];
        #pragma unroll
        for (int t = 0; t < 2; ++t) {
            const int n0 = (wave * 2 + t) * 16;
            const float bv = b1[m * H1 + n0 + l16];
            v8f acc;
            #pragma unroll
            for (int j = 0; j < 8; ++j) acc[j] = bv;
            const _Float16* wp = W1t + (size_t)m * N_F * H1 + (size_t)(n0 + l16) * N_F;
            #pragma unroll
            for (int ks = 0; ks < 8; ++ks) {
                const int k0 = ks * 32;
                v8h alo = *(const v8h*)&sA[l16 * N_F + k0 + 8 * hi];
                v8h ahi = *(const v8h*)&sA[l16 * N_F + k0 + 16 + 8 * hi];
                v8h blo = *(const v8h*)(wp + k0 + 8 * hi);
                v8h bhi = *(const v8h*)(wp + k0 + 16 + 8 * hi);
                v16h a, b;
                #pragma unroll
                for (int j = 0; j < 8; ++j) {
                    a[j] = alo[j]; a[8 + j] = ahi[j];
                    b[j] = blo[j]; b[8 + j] = bhi[j];
                }
                acc = __builtin_amdgcn_wmma_f32_16x16x32_f16(
                        false, a, false, b, (short)0, acc, false, false);
            }
            accL1[t] = acc;
        }
        __syncthreads();   // all waves done reading sA (layer-1 operands)

        // relu(a1) -> f16 back into sA (C layout: lane=col, vgpr v = row v+8*hi)
        #pragma unroll
        for (int t = 0; t < 2; ++t) {
            const int n0 = (wave * 2 + t) * 16;
            #pragma unroll
            for (int v = 0; v < 8; ++v) {
                float x = accL1[t][v];
                x = x > 0.f ? x : 0.f;
                sA[(v + 8 * hi) * N_F + n0 + l16] = (_Float16)x;
            }
        }
        __syncthreads();

        // ---- layer 2: one 16x16 output tile per wave; fold in W3 ----
        {
            const int n0 = wave * 16;
            const float bv = b2[m * H2 + n0 + l16];
            v8f acc;
            #pragma unroll
            for (int j = 0; j < 8; ++j) acc[j] = bv;
            const _Float16* wp = W2t + (size_t)m * H1 * H2 + (size_t)(n0 + l16) * H1;
            #pragma unroll
            for (int ks = 0; ks < 8; ++ks) {
                const int k0 = ks * 32;
                v8h alo = *(const v8h*)&sA[l16 * N_F + k0 + 8 * hi];
                v8h ahi = *(const v8h*)&sA[l16 * N_F + k0 + 16 + 8 * hi];
                v8h blo = *(const v8h*)(wp + k0 + 8 * hi);
                v8h bhi = *(const v8h*)(wp + k0 + 16 + 8 * hi);
                v16h a, b;
                #pragma unroll
                for (int j = 0; j < 8; ++j) {
                    a[j] = alo[j]; a[8 + j] = ahi[j];
                    b[j] = blo[j]; b[8 + j] = bhi[j];
                }
                acc = __builtin_amdgcn_wmma_f32_16x16x32_f16(
                        false, a, false, b, (short)0, acc, false, false);
            }
            const float w3v = W3[m * H2 + n0 + l16];
            #pragma unroll
            for (int v = 0; v < 8; ++v) {
                float x = acc[v];
                x = x > 0.f ? x : 0.f;
                sP[(v + 8 * hi) * H2 + n0 + l16] = x * w3v;
            }
        }
        __syncthreads();

        // ---- layer 3: reduce 128 cols per row, sigmoid, blend, store ----
        {
            const int r  = tid >> 4;
            const int c0 = (tid & 15) * 8;
            float s = 0.f;
            #pragma unroll
            for (int j = 0; j < 8; ++j) s += sP[r * H2 + c0 + j];
            sRow[r][tid & 15] = s;
        }
        __syncthreads();
        if (tid < 16) {
            float dot = b3[m];
            #pragma unroll
            for (int j = 0; j < 16; ++j) dot += sRow[tid][j];
            const float ratio = 1.0f / (1.0f + __expf(-dot));
            const size_t idx  = ((size_t)(b0 + tid)) * M_L + m;
            const float fw    = last_free_water[idx] + sFW[tid];
            const float fl    = fw * ratio + last_flow[idx] * (1.0f - ratio);
            out[idx] = fw - fl;                       // new_free_water
            out[(size_t)B_TOTAL * M_L + idx] = fl;    // flow
        }
        __syncthreads();   // protect sA/sRow reuse in next m iteration
    }
}

// ---------------------------------------------------------------------------
extern "C" void kernel_launch(void* const* d_in, const int* in_sizes, int n_in,
                              void* d_out, int out_size, void* d_ws, size_t ws_size,
                              hipStream_t stream) {
    const float* last_flow       = (const float*)d_in[0];
    const float* last_free_water = (const float*)d_in[1];
    const float* runoff          = (const float*)d_in[2];
    const float* W1              = (const float*)d_in[3];
    const float* b1              = (const float*)d_in[4];
    const float* W2              = (const float*)d_in[5];
    const float* b2              = (const float*)d_in[6];
    const float* W3              = (const float*)d_in[7];
    const float* b3              = (const float*)d_in[8];

    _Float16* W1t = (_Float16*)d_ws;                       // 384 KB
    _Float16* W2t = W1t + (size_t)M_L * N_F * H1;          // 192 KB

    const int total = M_L * N_F * H1 + M_L * H1 * H2;
    prep_weights<<<(total + 255) / 256, 256, 0, stream>>>(W1, W2, W1t, W2t);

    reservoir_kernel<<<B_TOTAL / 16, 256, 0, stream>>>(
        last_flow, last_free_water, runoff, b1, b2, W3, b3, W1t, W2t,
        (float*)d_out);
}
